// CausalSelfAttention_2327872275106
// MI455X (gfx1250) — matrix operand order repaired and checked by reading, not compiled, not run on hardware
//
#include <hip/hip_runtime.h>
#include <math.h>

// Problem constants (match reference)
static constexpr int BB = 4;
static constexpr int TT = 2048;
static constexpr int CC = 2048;
static constexpr int NHEADS = 16;
static constexpr int DH = 128;   // head dim

typedef __attribute__((ext_vector_type(16))) __bf16 v16bf;
typedef __attribute__((ext_vector_type(8)))  float  v8f;
typedef __attribute__((ext_vector_type(4)))  unsigned int u32x4;

union Frag {
  v16bf v;
  u32x4 q[2];
  unsigned short u[16];
};

__device__ __forceinline__ unsigned short f2bf(float f) {
  unsigned u = __float_as_uint(f);
  unsigned r = u + 0x7FFFu + ((u >> 16) & 1u);   // round-to-nearest-even
  return (unsigned short)(r >> 16);
}

// ---------------- prep kernels ----------------

__global__ void k_f32_to_bf16(const float* __restrict__ src,
                              unsigned short* __restrict__ dst, long n) {
  long i = (long)blockIdx.x * blockDim.x + threadIdx.x;
  if (i < n) dst[i] = f2bf(src[i]);
}

// src [K][N] f32 -> dst [N][K] bf16 (transpose + convert)
__global__ void k_transpose_bf16(const float* __restrict__ src,
                                 unsigned short* __restrict__ dst,
                                 int K, int N) {
  long i = (long)blockIdx.x * blockDim.x + threadIdx.x;
  long total = (long)K * N;
  if (i >= total) return;
  int n = (int)(i / K);
  int k = (int)(i % K);
  dst[i] = f2bf(src[(long)k * N + n]);
}

// ---------------- bf16 WMMA GEMM: C[M][N] = A[M][K] * BT[N][K]^T ----------------
// 8 waves/block; wave computes 32x32 via 2x2 v_wmma_f32_16x16x32_bf16 tiles.
// Block tile 64(M) x 128(N). Operands are L2-resident (192MB), fragments loaded
// directly from global with contiguous 16B chunks (k = kh*16 + j).

__global__ __launch_bounds__(256)
void k_gemm_bf16(const unsigned short* __restrict__ A,   // [M][K] bf16
                 const unsigned short* __restrict__ BT,  // [N][K] bf16
                 float* __restrict__ Cout,               // [M][N] f32
                 int M, int N, int K) {
  const int wv    = threadIdx.x >> 5;
  const int lane  = threadIdx.x & 31;
  const int laneN = lane & 15;
  const int kh    = lane >> 4;
  const int wm    = wv & 1;
  const int wn    = wv >> 1;
  const int m0 = blockIdx.y * 64  + wm * 32;
  const int n0 = blockIdx.x * 128 + wn * 32;

  const unsigned short* A0 = A  + (size_t)(m0 + laneN)      * K;
  const unsigned short* A1 = A  + (size_t)(m0 + 16 + laneN) * K;
  const unsigned short* B0 = BT + (size_t)(n0 + laneN)      * K;
  const unsigned short* B1 = BT + (size_t)(n0 + 16 + laneN) * K;

  v8f c00 = {}, c01 = {}, c10 = {}, c11 = {};

  for (int k0 = 0; k0 < K; k0 += 32) {
    Frag a0, a1, b0, b1;
    // A fragment (16x32 bf16): lane m=laneN; elems: K = kh*16 + j (contiguous 16)
    a0.q[0] = *(const u32x4*)(A0 + k0 + kh * 16);
    a0.q[1] = *(const u32x4*)(A0 + k0 + kh * 16 + 8);
    a1.q[0] = *(const u32x4*)(A1 + k0 + kh * 16);
    a1.q[1] = *(const u32x4*)(A1 + k0 + kh * 16 + 8);
    // B fragment (32x16 bf16): lane n=laneN; elems: K = kh*16 + j (contiguous 16)
    b0.q[0] = *(const u32x4*)(B0 + k0 + kh * 16);
    b0.q[1] = *(const u32x4*)(B0 + k0 + kh * 16 + 8);
    b1.q[0] = *(const u32x4*)(B1 + k0 + kh * 16);
    b1.q[1] = *(const u32x4*)(B1 + k0 + kh * 16 + 8);

    if (k0 + 32 < K) {
      __builtin_prefetch(A0 + k0 + 32, 0, 3);   // global_prefetch_b8
      __builtin_prefetch(B0 + k0 + 32, 0, 3);
    }

    c00 = __builtin_amdgcn_wmma_f32_16x16x32_bf16(false, a0.v, false, b0.v, (short)0, c00, false, false);
    c01 = __builtin_amdgcn_wmma_f32_16x16x32_bf16(false, a0.v, false, b1.v, (short)0, c01, false, false);
    c10 = __builtin_amdgcn_wmma_f32_16x16x32_bf16(false, a1.v, false, b0.v, (short)0, c10, false, false);
    c11 = __builtin_amdgcn_wmma_f32_16x16x32_bf16(false, a1.v, false, b1.v, (short)0, c11, false, false);
  }

  // C layout: lanes 0-15 rows 0-7, lanes 16-31 rows 8-15; col = laneN
  #pragma unroll
  for (int r = 0; r < 8; ++r) {
    int row0 = m0 + kh * 8 + r;
    int row1 = row0 + 16;
    float* p0 = Cout + (size_t)row0 * N + n0 + laneN;
    float* p1 = Cout + (size_t)row1 * N + n0 + laneN;
    p0[0]  = c00[r];
    p0[16] = c01[r];
    p1[0]  = c10[r];
    p1[16] = c11[r];
  }
}

// ---------------- RoPE + repack: qkv f32 -> Q/K [bh][t][128] bf16, V^T [bh][128][t] bf16 ----

__global__ __launch_bounds__(128)
void k_rope_pack(const float* __restrict__ qkv,          // [B][T][3C]
                 unsigned short* __restrict__ Qh,        // [B*NH][T][DH]
                 unsigned short* __restrict__ Kh,        // [B*NH][T][DH]
                 unsigned short* __restrict__ VTr) {     // [B*NH][DH][T]
  const int t  = blockIdx.x;
  const int bh = blockIdx.y;
  const int b  = bh / NHEADS;
  const int h  = bh % NHEADS;
  const int d  = threadIdx.x;                            // 0..127
  const size_t base = ((size_t)(b * TT + t)) * (3 * CC) + h * DH;

  float qv = qkv[base + d];
  float kv = qkv[base + CC + d];
  float vv = qkv[base + 2 * CC + d];

  int dr = (d < 64) ? d : d - 64;
  float inv = __expf(-(float)dr * 0.14391156507f);       // 10000^(-dr/64)
  float ang = (float)t * inv;
  float cs = __cosf(ang), sn = __sinf(ang);

  if (d < 64) {
    float q2 = qkv[base + d + 64];
    float k2 = qkv[base + CC + d + 64];
    qv = qv * cs + q2 * sn;
    kv = kv * cs + k2 * sn;
  } else {
    float q1 = qkv[base + d - 64];
    float k1 = qkv[base + CC + d - 64];
    qv = qv * cs - q1 * sn;
    kv = kv * cs - k1 * sn;
  }

  size_t o = ((size_t)bh * TT + t) * DH + d;
  Qh[o] = f2bf(qv);
  Kh[o] = f2bf(kv);
  VTr[((size_t)bh * DH + d) * TT + t] = f2bf(vv);
}

// ---------------- flash attention: one wave per 16-row Q tile -------------------
// S(16x32) via 8 WMMAs, online softmax (shfl_xor width-16 row reductions),
// P relaid C->A layout through per-wave LDS, PV via 8 WMMAs (V^T gives
// contiguous B fragments). Output written as bf16 for the out-proj GEMM.

__global__ __launch_bounds__(256)
void k_flash(const unsigned short* __restrict__ Qh,
             const unsigned short* __restrict__ Kh,
             const unsigned short* __restrict__ VTr,
             unsigned short* __restrict__ Ybf) {         // [B][T][C] bf16
  __shared__ __align__(16) unsigned short pbuf[8 * 16 * 32];  // 8 waves x (16x32 bf16)

  const int wv    = threadIdx.x >> 5;
  const int lane  = threadIdx.x & 31;
  const int laneN = lane & 15;
  const int kh    = lane >> 4;
  const int qt    = blockIdx.x * 8 + wv;
  const int q0    = qt * 16;
  const int bh    = blockIdx.y;
  const int b     = bh / NHEADS;
  const int h     = bh % NHEADS;

  const unsigned short* Q = Qh  + (size_t)bh * TT * DH;
  const unsigned short* K = Kh  + (size_t)bh * TT * DH;
  const unsigned short* V = VTr + (size_t)bh * DH * TT;

  // Q fragments for all 4 k-steps of head_dim=128 (loaded once)
  Frag qf[4];
  const unsigned short* qrow = Q + (size_t)(q0 + laneN) * DH;
  #pragma unroll
  for (int s = 0; s < 4; ++s) {
    qf[s].q[0] = *(const u32x4*)(qrow + s * 32 + kh * 16);
    qf[s].q[1] = *(const u32x4*)(qrow + s * 32 + kh * 16 + 8);
  }

  float mi[8], li[8];
  v8f o[8];
  v8f vzero = {};
  #pragma unroll
  for (int r = 0; r < 8; ++r) { mi[r] = -__builtin_inff(); li[r] = 0.f; }
  #pragma unroll
  for (int t = 0; t < 8; ++t) o[t] = vzero;

  const float scale = 0.08838834764831845f;              // 1/sqrt(128)
  unsigned short* lds = pbuf + wv * 512;                 // 16x32 bf16 per wave
  const int jmax = (q0 + 47) >> 5;                       // causal key-tile bound

  for (int j = 0; j < jmax; ++j) {
    const int key0 = j * 32;
    v8f sA = vzero, sB = vzero;
    const unsigned short* k0row = K + (size_t)(key0 + laneN) * DH;
    const unsigned short* k1row = K + (size_t)(key0 + 16 + laneN) * DH;
    #pragma unroll
    for (int s = 0; s < 4; ++s) {
      Frag bfa, bfb;
      bfa.q[0] = *(const u32x4*)(k0row + s * 32 + kh * 16);
      bfa.q[1] = *(const u32x4*)(k0row + s * 32 + kh * 16 + 8);
      bfb.q[0] = *(const u32x4*)(k1row + s * 32 + kh * 16);
      bfb.q[1] = *(const u32x4*)(k1row + s * 32 + kh * 16 + 8);
      sA = __builtin_amdgcn_wmma_f32_16x16x32_bf16(false, qf[s].v, false, bfa.v, (short)0, sA, false, false);
      sB = __builtin_amdgcn_wmma_f32_16x16x32_bf16(false, qf[s].v, false, bfb.v, (short)0, sB, false, false);
    }

    // scale + causal mask + online softmax update, row by row
    const int keyA = key0 + laneN;
    const int keyB = key0 + 16 + laneN;
    #pragma unroll
    for (int r = 0; r < 8; ++r) {
      const int qrowi = q0 + kh * 8 + r;
      float a = sA[r] * scale;
      float c = sB[r] * scale;
      if (keyA > qrowi) a = -__builtin_inff();
      if (keyB > qrowi) c = -__builtin_inff();
      float tmax = fmaxf(a, c);
      #pragma unroll
      for (int m = 1; m < 16; m <<= 1) tmax = fmaxf(tmax, __shfl_xor(tmax, m, 16));
      float nm    = fmaxf(mi[r], tmax);
      float alpha = __expf(mi[r] - nm);
      float pa    = __expf(a - nm);
      float pc    = __expf(c - nm);
      float rs = pa + pc;
      #pragma unroll
      for (int m = 1; m < 16; m <<= 1) rs += __shfl_xor(rs, m, 16);
      li[r] = li[r] * alpha + rs;
      mi[r] = nm;
      #pragma unroll
      for (int t = 0; t < 8; ++t) o[t][r] = o[t][r] * alpha;
      // stage P in LDS (C-layout -> row-major 16x32 tile)
      lds[(kh * 8 + r) * 32 + laneN]      = f2bf(pa);
      lds[(kh * 8 + r) * 32 + 16 + laneN] = f2bf(pc);
    }

    // same-wave DS ops are in-order; fence so the A-fragment reads see the stores
    asm volatile("s_wait_dscnt 0" ::: "memory");

    Frag pf;
    pf.q[0] = *(const u32x4*)(lds + laneN * 32 + kh * 16);
    pf.q[1] = *(const u32x4*)(lds + laneN * 32 + kh * 16 + 8);

    // O(16x128) += P(16x32) x V(32x128); V^T rows give contiguous B fragments
    #pragma unroll
    for (int t = 0; t < 8; ++t) {
      const unsigned short* vrow = V + (size_t)(t * 16 + laneN) * TT + key0 + kh * 16;
      Frag vf;
      vf.q[0] = *(const u32x4*)(vrow);
      vf.q[1] = *(const u32x4*)(vrow + 8);
      o[t] = __builtin_amdgcn_wmma_f32_16x16x32_bf16(false, pf.v, false, vf.v, (short)0, o[t], false, false);
    }
  }

  // normalize and emit bf16 y[b][t][h*128 + d]
  #pragma unroll
  for (int r = 0; r < 8; ++r) {
    float inv = 1.0f / li[r];
    int row = q0 + kh * 8 + r;
    unsigned short* yrow = Ybf + ((size_t)(b * TT + row)) * CC + h * DH;
    #pragma unroll
    for (int t = 0; t < 8; ++t)
      yrow[t * 16 + laneN] = f2bf(o[t][r] * inv);
  }
}

// ---------------- launch ----------------

extern "C" void kernel_launch(void* const* d_in, const int* in_sizes, int n_in,
                              void* d_out, int out_size, void* d_ws, size_t ws_size,
                              hipStream_t stream) {
  const float* x      = (const float*)d_in[0];
  const float* w_attn = (const float*)d_in[1];
  const float* w_proj = (const float*)d_in[2];
  float* out = (float*)d_out;
  char* ws = (char*)d_ws;

  const int M = BB * TT;  // 8192

  // workspace layout (384 MB total, 256B-aligned offsets)
  unsigned short* x_bf = (unsigned short*)(ws + 0);           //  33.6 MB  [M][C] bf16
  unsigned short* waT  = (unsigned short*)(ws + 33554432);    //  25.2 MB  [3C][C] bf16
  unsigned short* wpT  = (unsigned short*)(ws + 58720256);    //   8.4 MB  [C][C] bf16
  float*          qkv  = (float*)        (ws + 67108864);     // 201.3 MB  [M][3C] f32
  unsigned short* Qh   = (unsigned short*)(ws + 268435456);   //  33.6 MB
  unsigned short* Kh   = (unsigned short*)(ws + 301989888);   //  33.6 MB
  unsigned short* VTr  = (unsigned short*)(ws + 335544320);   //  33.6 MB
  unsigned short* Ybf  = (unsigned short*)(ws + 369098752);   //  33.6 MB

  {
    long n = (long)M * CC;
    k_f32_to_bf16<<<dim3((unsigned)((n + 255) / 256)), dim3(256), 0, stream>>>(x, x_bf, n);
  }
  {
    long n = (long)3 * CC * CC;
    k_transpose_bf16<<<dim3((unsigned)((n + 255) / 256)), dim3(256), 0, stream>>>(w_attn, waT, CC, 3 * CC);
  }
  {
    long n = (long)CC * CC;
    k_transpose_bf16<<<dim3((unsigned)((n + 255) / 256)), dim3(256), 0, stream>>>(w_proj, wpT, CC, CC);
  }

  // qkv = x @ w_attn
  k_gemm_bf16<<<dim3(3 * CC / 128, M / 64), dim3(256), 0, stream>>>(x_bf, waT, qkv, M, 3 * CC, CC);

  // RoPE + per-head repack (V transposed)
  k_rope_pack<<<dim3(TT, BB * NHEADS), dim3(128), 0, stream>>>(qkv, Qh, Kh, VTr);

  // flash attention -> y (bf16)
  k_flash<<<dim3(TT / 16 / 8, BB * NHEADS), dim3(256), 0, stream>>>(Qh, Kh, VTr, Ybf);

  // out = y @ w_proj (f32 output)
  k_gemm_bf16<<<dim3(CC / 128, M / 64), dim3(256), 0, stream>>>(Ybf, wpT, out, M, CC, CC);
}
